// IntegratedDCABlock_45071386804651
// MI455X (gfx1250) — matrix operand order repaired and checked by reading, not compiled, not run on hardware
//
#include <hip/hip_runtime.h>
#include <hip/hip_bf16.h>

#define DIM   1024
#define NH    16
#define HDIM  64
#define CR    4
#define BATCH 4
#define SEQ   1024
#define PAST  4096
#define CLEN  (PAST / CR)     // 1024 compressed keys
#define BS    (BATCH * SEQ)   // 4096 rows
#define ATT_SCALE 0.125f      // 1/sqrt(64)

typedef __attribute__((ext_vector_type(16))) __bf16 bf16x16;
typedef __attribute__((ext_vector_type(8)))  float  f32x8;
typedef __attribute__((__vector_size__(16))) int    vi4;

union BF16Frag {
  bf16x16 v;
  unsigned int   u32[8];
  unsigned short u16[16];
};
union Pack8 {           // 8 ushorts <-> one uint4 (16B)
  uint4 q;
  unsigned int   u32[4];
  unsigned short u16[8];
};

__device__ __forceinline__ unsigned short f2bf(float f) {
  unsigned int u = __float_as_uint(f);
  if ((u & 0x7f800000u) == 0x7f800000u) return (unsigned short)(u >> 16); // inf/nan
  u += 0x7fffu + ((u >> 16) & 1u);  // round-to-nearest-even
  return (unsigned short)(u >> 16);
}

__device__ __forceinline__ f32x8 wmma_bf16(bf16x16 a, bf16x16 b, f32x8 c) {
  return __builtin_amdgcn_wmma_f32_16x16x32_bf16(false, a, false, b, (short)0, c,
                                                 false, false);
}

// ---- CDNA5 async global->LDS copy (GLOBAL_LOAD_ASYNC_TO_LDS_B128) ----
// Builtin signature (from clang diagnostic): (v4i addrspace(1)*, v4i addrspace(3)*,
// imm int offset, imm int cpol)
#if __has_builtin(__builtin_amdgcn_global_load_async_to_lds_b128)
#define ASYNC_LDS 1
__device__ __forceinline__ void async_copy16(const void* g, void* l) {
  __builtin_amdgcn_global_load_async_to_lds_b128(
      (__attribute__((address_space(1))) vi4*)g,
      (__attribute__((address_space(3))) vi4*)l, 0, 0);
}
__device__ __forceinline__ void wait_async() {
#if __has_builtin(__builtin_amdgcn_s_wait_asynccnt)
  __builtin_amdgcn_s_wait_asynccnt(0);
#else
  asm volatile("s_wait_asynccnt 0x0" ::: "memory");
#endif
  asm volatile("" ::: "memory");
}
#else
#define ASYNC_LDS 0
#endif

// Branchless exact-GELU (erf via Abramowitz-Stegun 7.1.26, |err|<1.5e-7)
__device__ __forceinline__ float gelu_exact(float x) {
  float z  = x * 0.70710678118f;
  float az = fabsf(z);
  float t  = 1.f / fmaf(0.3275911f, az, 1.f);
  float p  = t * fmaf(t, fmaf(t, fmaf(t, fmaf(t, 1.061405429f, -1.453152027f),
                                      1.421413741f), -0.284496736f), 0.254829592f);
  float e  = 1.f - p * __expf(-az * az);
  float erfz = (z < 0.f) ? -e : e;
  return 0.5f * x * (1.f + erfz);
}

// ---------------------------------------------------------------- cvt f32->bf16
__global__ void cvt_kernel(const float* __restrict__ in,
                           unsigned short* __restrict__ out, int n) {
  int i = blockIdx.x * blockDim.x + threadIdx.x;
  if (i < n) out[i] = f2bf(in[i]);
}

// ---------------------------------------------------------------- LayerNorm
__global__ __launch_bounds__(256) void ln_kernel(
    const float* __restrict__ x, const float* __restrict__ g,
    const float* __restrict__ bt, float* __restrict__ of,
    unsigned short* __restrict__ ob) {
  __shared__ float sbuf[16];
  int row = blockIdx.x;
  const float* xr = x + (size_t)row * DIM;
  int tid = threadIdx.x, lane = tid & 31, wid = tid >> 5;
  float s = 0.f, s2 = 0.f;
  for (int i = tid; i < DIM; i += 256) { float v = xr[i]; s += v; s2 += v * v; }
  #pragma unroll
  for (int o = 16; o > 0; o >>= 1) { s += __shfl_xor(s, o); s2 += __shfl_xor(s2, o); }
  if (lane == 0) { sbuf[wid] = s; sbuf[8 + wid] = s2; }
  __syncthreads();
  if (tid == 0) {
    float a = 0.f, b2 = 0.f;
    for (int i = 0; i < 8; ++i) { a += sbuf[i]; b2 += sbuf[8 + i]; }
    sbuf[0] = a; sbuf[8] = b2;
  }
  __syncthreads();
  float mu  = sbuf[0] / DIM;
  float var = sbuf[8] / DIM - mu * mu;
  float rstd = rsqrtf(var + 1e-5f);
  for (int i = tid; i < DIM; i += 256) {
    float y = (xr[i] - mu) * rstd * g[i] + bt[i];
    if (of) of[(size_t)row * DIM + i] = y;
    if (ob) ob[(size_t)row * DIM + i] = f2bf(y);
  }
}

// ---------------------------------------------------------------- mean over seq
__global__ void colmean_kernel(const float* __restrict__ x, float* __restrict__ out) {
  int d = blockIdx.x * blockDim.x + threadIdx.x;
  int b = blockIdx.y;
  const float* p = x + (size_t)b * SEQ * DIM + d;
  float s = 0.f;
  for (int t = 0; t < SEQ; ++t) s += p[(size_t)t * DIM];
  out[b * DIM + d] = s / (float)SEQ;
}

// ---------------------------------------------------------------- 2-way gate
__global__ __launch_bounds__(256) void gate_kernel(
    const float* __restrict__ mean, const float* __restrict__ Wg,
    const float* __restrict__ bg, float* __restrict__ gate) {
  __shared__ float r0[256], r1[256];
  int b = blockIdx.x, tid = threadIdx.x;
  float s0 = 0.f, s1 = 0.f;
  for (int d = tid; d < DIM; d += 256) {
    float m = mean[b * DIM + d];
    s0 += m * Wg[d * 2];
    s1 += m * Wg[d * 2 + 1];
  }
  r0[tid] = s0; r1[tid] = s1;
  __syncthreads();
  for (int o = 128; o > 0; o >>= 1) {
    if (tid < o) { r0[tid] += r0[tid + o]; r1[tid] += r1[tid + o]; }
    __syncthreads();
  }
  if (tid == 0) {
    float l0 = r0[0] + bg[0], l1 = r1[0] + bg[1];
    float mx = fmaxf(l0, l1);
    float e0 = __expf(l0 - mx), e1 = __expf(l1 - mx);
    float inv = 1.f / (e0 + e1);
    gate[b * 2] = e0 * inv;
    gate[b * 2 + 1] = e1 * inv;
  }
}

// ---------------------------------------------------------------- CR mean-pool
__global__ void pool_kernel(const float* __restrict__ past,
                            unsigned short* __restrict__ out, int n) {
  int i = blockIdx.x * blockDim.x + threadIdx.x;
  if (i >= n) return;
  int d  = i % DIM;
  int rc = (i / DIM) % CLEN;
  int b  = i / (DIM * CLEN);
  const float* p = past + ((size_t)(b * PAST + rc * CR)) * DIM + d;
  float s = 0.f;
  #pragma unroll
  for (int j = 0; j < CR; ++j) s += p[(size_t)j * DIM];
  out[i] = f2bf(s * (1.f / CR));
}

// ---------------------------------------------------------------- gated combine
__global__ void combine_kernel(const float* __restrict__ o0,
                               const float* __restrict__ o1,
                               const float* __restrict__ gate,
                               unsigned short* __restrict__ out, int n) {
  int i = blockIdx.x * blockDim.x + threadIdx.x;
  if (i >= n) return;
  int b = i / (SEQ * DIM);
  out[i] = f2bf(gate[b * 2] * o0[i] + gate[b * 2 + 1] * o1[i]);
}

// ---------------------------------------------------------------- WMMA GEMM
// C[M,N] = act(A[M,K](bf16) * W[K,N](bf16) + bias) (+ residual)
// N,K,ACT,RES,OUTF compile-time: immediate-offset loads, branch-free epilogue.
// Block: 256 thr (8 waves), 128x64 tile; wave = 32x32 patch = 4 WMMAs/K-step.
template <int N, int K, int ACT, bool RES, bool OUTF>
__global__ __launch_bounds__(256) void gemm_bf16_kernel(
    const unsigned short* __restrict__ A, const unsigned short* __restrict__ W,
    const float* __restrict__ bias, void* __restrict__ Cout,
    const float* __restrict__ resid, int M) {
  __shared__ __align__(16) unsigned short lA[128 * 32];   // [m][k]
  __shared__ __align__(16) unsigned short lBt[64 * 32];   // [n][k]  (transposed)
  int tid = threadIdx.x;
  int lane = tid & 31, w = tid >> 5;
  int half = lane >> 4;
  int col  = lane & 15;
  int m0 = blockIdx.y * 128, n0 = blockIdx.x * 64;
  int mt = (w & 3) * 32;
  int nt = (w >> 2) * 32;

  f32x8 c00, c01, c10, c11;
  #pragma unroll
  for (int i = 0; i < 8; ++i) { c00[i] = 0.f; c01[i] = 0.f; c10[i] = 0.f; c11[i] = 0.f; }

  int ar = tid >> 1;           // A tile: 128 rows, thread covers 16 ushorts
  int ac = (tid & 1) * 16;
  int bn = tid & 63;           // W tile: thread owns 8 k's at one n
  int bk = (tid >> 6) * 8;
  const unsigned short* ga0 = &A[(size_t)(m0 + ar) * K + ac];
  const unsigned short* gw0 = &W[(size_t)bk * N + n0 + bn];

  for (int k0 = 0; k0 < K; k0 += 32) {
    // ---- stage A (row-major), async global->LDS when available ----
#if ASYNC_LDS
    async_copy16(ga0 + k0,     &lA[ar * 32 + ac]);
    async_copy16(ga0 + k0 + 8, &lA[ar * 32 + ac + 8]);
#else
    *(uint4*)&lA[ar * 32 + ac]     = *(const uint4*)(ga0 + k0);
    *(uint4*)&lA[ar * 32 + ac + 8] = *(const uint4*)(ga0 + k0 + 8);
#endif
    // ---- stage W transposed: 8 immediate-offset u16 loads -> 1 b128 store ----
    {
      Pack8 pk;
      const unsigned short* gw = gw0 + (size_t)k0 * N;
      #pragma unroll
      for (int j = 0; j < 8; ++j) pk.u16[j] = gw[j * N];
      *(uint4*)&lBt[bn * 32 + bk] = pk.q;
    }
    if (k0 + 32 < K) {
      __builtin_prefetch(ga0 + k0 + 32, 0, 3);
      __builtin_prefetch(gw0 + (size_t)(k0 + 32) * N, 0, 3);
    }
#if ASYNC_LDS
    wait_async();
#endif
    __syncthreads();

    BF16Frag fa0, fa1, fb0, fb1;
    #pragma unroll
    for (int vg = 0; vg < 8; ++vg) {  // ISA A-layout 16x32 (16-bit)
      int kk = ((vg & 4) ? 16 : 0) + half * 8 + (vg & 3) * 2;
      fa0.u32[vg] = *(const unsigned int*)&lA[(mt + col) * 32 + kk];
      fa1.u32[vg] = *(const unsigned int*)&lA[(mt + 16 + col) * 32 + kk];
    }
    #pragma unroll
    for (int vg = 0; vg < 8; ++vg) {  // ISA B-layout 32x16, from transposed LDS
      int kk = ((vg & 4) ? 16 : 0) + half * 8 + (vg & 3) * 2;
      fb0.u32[vg] = *(const unsigned int*)&lBt[(nt + col) * 32 + kk];
      fb1.u32[vg] = *(const unsigned int*)&lBt[(nt + 16 + col) * 32 + kk];
    }
    c00 = wmma_bf16(fa0.v, fb0.v, c00);
    c01 = wmma_bf16(fa0.v, fb1.v, c01);
    c10 = wmma_bf16(fa1.v, fb0.v, c10);
    c11 = wmma_bf16(fa1.v, fb1.v, c11);
    __syncthreads();
  }

  #pragma unroll
  for (int mi = 0; mi < 2; ++mi) {
    #pragma unroll
    for (int ni = 0; ni < 2; ++ni) {
      f32x8 cc = mi ? (ni ? c11 : c10) : (ni ? c01 : c00);
      int n = n0 + nt + ni * 16 + col;
      float bv = bias[n];
      #pragma unroll
      for (int v = 0; v < 8; ++v) {
        int m = m0 + mt + mi * 16 + v + 8 * half;   // C-layout row mapping
        float x = cc[v] + bv;
        if (ACT == 1) x = gelu_exact(x);
        size_t o = (size_t)m * N + n;
        if (RES) x += resid[o];
        if (OUTF) ((float*)Cout)[o] = x;
        else      ((unsigned short*)Cout)[o] = f2bf(x);
      }
    }
  }
}

// ---------------------------------------------------------------- flash attention
// Q,K,V as (b, s, NH*HDIM) bf16, Skv = 1024 for both context branches.
// Block 128 thr (4 waves); each wave owns 16 q-rows, streams 32-key tiles.
// V tile staged TRANSPOSED ([hd][key]); K tile async-copied to LDS.
template <int CAUSAL>
__global__ __launch_bounds__(128) void flash_attn_kernel(
    const unsigned short* __restrict__ Q, const unsigned short* __restrict__ Kb,
    const unsigned short* __restrict__ Vb, float* __restrict__ Out) {
  const int Skv = 1024;
  __shared__ __align__(16) unsigned short lds[4 * 4608]; // per wave: K2048 Vt2048 P512
  int tid = threadIdx.x;
  int lane = tid & 31, w = tid >> 5;
  int half = lane >> 4;
  int col = lane & 15;
  unsigned short* lK  = lds + w * 4608;
  unsigned short* lVt = lK + 2048;
  unsigned short* lP  = lVt + 2048;
  int b = blockIdx.y / NH, h = blockIdx.y % NH;
  int q0 = blockIdx.x * 64 + w * 16;

  // Q fragments (A-layout), K-dim (HD=64) split into two 32-chunks
  BF16Frag aq0, aq1;
  {
    size_t qbase = ((size_t)(b * SEQ + q0 + col)) * DIM + h * HDIM;
    #pragma unroll
    for (int vg = 0; vg < 8; ++vg) {
      int kk = ((vg & 4) ? 16 : 0) + half * 8 + (vg & 3) * 2;
      aq0.u32[vg] = *(const unsigned int*)&Q[qbase + kk];
      aq1.u32[vg] = *(const unsigned int*)&Q[qbase + 32 + kk];
    }
  }

  f32x8 o0t, o1t, o2t, o3t;
  #pragma unroll
  for (int i = 0; i < 8; ++i) { o0t[i] = 0.f; o1t[i] = 0.f; o2t[i] = 0.f; o3t[i] = 0.f; }
  float mrow[8], lrow[8];
  #pragma unroll
  for (int v = 0; v < 8; ++v) { mrow[v] = -1e30f; lrow[v] = 0.f; }

  int ntiles = CAUSAL ? (q0 + 47) / 32 : (Skv / 32);
  for (int kt = 0; kt < ntiles; ++kt) {
    int k0 = kt * 32;
    { // K tile row-major (pairs along hd contiguous); async copy to LDS
      size_t gb = ((size_t)(b * Skv + k0 + lane)) * DIM + h * HDIM;
#if ASYNC_LDS
      #pragma unroll
      for (int i = 0; i < 8; ++i)
        async_copy16(&Kb[gb + i * 8], &lK[lane * 64 + i * 8]);
#else
      const uint4* gk = (const uint4*)&Kb[gb];
      uint4* dk = (uint4*)&lK[lane * 64];
      #pragma unroll
      for (int i = 0; i < 8; ++i) dk[i] = gk[i];
#endif
    }
    { // V tile transposed: lane owns hd pair (2*lane, 2*lane+1)
      size_t vb = ((size_t)(b * Skv + k0)) * DIM + h * HDIM + 2 * lane;
      #pragma unroll
      for (int kh = 0; kh < 2; ++kh) {
        unsigned int tmp[16];
        #pragma unroll
        for (int j = 0; j < 16; ++j)
          tmp[j] = *(const unsigned int*)&Vb[vb + (size_t)(kh * 16 + j) * DIM];
        Pack8 r0a, r0b, r1a, r1b;
        #pragma unroll
        for (int j = 0; j < 4; ++j) {
          unsigned int a = tmp[2 * j], c = tmp[2 * j + 1];
          r0a.u32[j] = (a & 0xffffu) | (c << 16);
          r1a.u32[j] = (a >> 16) | (c & 0xffff0000u);
          unsigned int a2 = tmp[8 + 2 * j], c2 = tmp[8 + 2 * j + 1];
          r0b.u32[j] = (a2 & 0xffffu) | (c2 << 16);
          r1b.u32[j] = (a2 >> 16) | (c2 & 0xffff0000u);
        }
        *(uint4*)&lVt[(2 * lane) * 32 + kh * 16]         = r0a.q;
        *(uint4*)&lVt[(2 * lane) * 32 + kh * 16 + 8]     = r0b.q;
        *(uint4*)&lVt[(2 * lane + 1) * 32 + kh * 16]     = r1a.q;
        *(uint4*)&lVt[(2 * lane + 1) * 32 + kh * 16 + 8] = r1b.q;
      }
    }
#if ASYNC_LDS
    wait_async();
#endif

    // scores: S = Q(16x64) * K^T(64x32) -> two 16x16 C tiles
    f32x8 s0, s1;
    #pragma unroll
    for (int i = 0; i < 8; ++i) { s0[i] = 0.f; s1[i] = 0.f; }
    {
      BF16Frag fb;
      #pragma unroll
      for (int vg = 0; vg < 8; ++vg)
        fb.u32[vg] = *(const unsigned int*)&lK[col * 64 + ((vg & 4) ? 16 : 0) + half * 8 + (vg & 3) * 2];
      s0 = wmma_bf16(aq0.v, fb.v, s0);
      #pragma unroll
      for (int vg = 0; vg < 8; ++vg)
        fb.u32[vg] = *(const unsigned int*)&lK[col * 64 + 32 + ((vg & 4) ? 16 : 0) + half * 8 + (vg & 3) * 2];
      s0 = wmma_bf16(aq1.v, fb.v, s0);
      #pragma unroll
      for (int vg = 0; vg < 8; ++vg)
        fb.u32[vg] = *(const unsigned int*)&lK[(16 + col) * 64 + ((vg & 4) ? 16 : 0) + half * 8 + (vg & 3) * 2];
      s1 = wmma_bf16(aq0.v, fb.v, s1);
      #pragma unroll
      for (int vg = 0; vg < 8; ++vg)
        fb.u32[vg] = *(const unsigned int*)&lK[(16 + col) * 64 + 32 + ((vg & 4) ? 16 : 0) + half * 8 + (vg & 3) * 2];
      s1 = wmma_bf16(aq1.v, fb.v, s1);
    }

    // online softmax (row reductions across 16-lane groups)
    bool needmask = CAUSAL && (k0 + 31 > q0);
    float p0[8], p1[8], alpha[8];
    #pragma unroll
    for (int v = 0; v < 8; ++v) {
      float a  = s0[v] * ATT_SCALE;
      float bb = s1[v] * ATT_SCALE;
      if (needmask) {
        int qrow = q0 + v + 8 * half;
        if (k0 + col > qrow) a = -1e30f;
        if (k0 + 16 + col > qrow) bb = -1e30f;
      }
      float mx = fmaxf(a, bb);
      #pragma unroll
      for (int o = 1; o < 16; o <<= 1) mx = fmaxf(mx, __shfl_xor(mx, o));
      float mn = fmaxf(mrow[v], mx);
      float al = __expf(mrow[v] - mn);
      float e0 = (a  < -1e29f) ? 0.f : __expf(a - mn);
      float e1 = (bb < -1e29f) ? 0.f : __expf(bb - mn);
      float rs = e0 + e1;
      #pragma unroll
      for (int o = 1; o < 16; o <<= 1) rs += __shfl_xor(rs, o);
      lrow[v] = lrow[v] * al + rs;
      mrow[v] = mn;
      alpha[v] = al;
      p0[v] = e0; p1[v] = e1;
    }
    #pragma unroll
    for (int v = 0; v < 8; ++v) {
      o0t[v] *= alpha[v]; o1t[v] *= alpha[v]; o2t[v] *= alpha[v]; o3t[v] *= alpha[v];
      int r = v + 8 * half;                 // C-layout -> LDS (16x32) re-layout
      lP[r * 32 + col]      = f2bf(p0[v]);
      lP[r * 32 + 16 + col] = f2bf(p1[v]);
    }

    // P (16x32) as A fragment; V^T tiles as B fragments (pairs contiguous)
    BF16Frag fp;
    #pragma unroll
    for (int vg = 0; vg < 8; ++vg) {
      int kk = ((vg & 4) ? 16 : 0) + half * 8 + (vg & 3) * 2;
      fp.u32[vg] = *(const unsigned int*)&lP[col * 32 + kk];
    }
    #pragma unroll
    for (int t = 0; t < 4; ++t) {
      BF16Frag fb;
      #pragma unroll
      for (int vg = 0; vg < 8; ++vg) {
        int kk = ((vg & 4) ? 16 : 0) + half * 8 + (vg & 3) * 2;
        fb.u32[vg] = *(const unsigned int*)&lVt[(t * 16 + col) * 32 + kk];
      }
      if (t == 0) o0t = wmma_bf16(fp.v, fb.v, o0t);
      else if (t == 1) o1t = wmma_bf16(fp.v, fb.v, o1t);
      else if (t == 2) o2t = wmma_bf16(fp.v, fb.v, o2t);
      else o3t = wmma_bf16(fp.v, fb.v, o3t);
    }
  }

  size_t obase = ((size_t)(b * SEQ + q0)) * DIM + h * HDIM;
  #pragma unroll
  for (int v = 0; v < 8; ++v) {
    float inv = lrow[v] > 0.f ? 1.f / lrow[v] : 0.f;
    size_t rb = obase + (size_t)(v + 8 * half) * DIM;
    Out[rb + col]      = o0t[v] * inv;
    Out[rb + 16 + col] = o1t[v] * inv;
    Out[rb + 32 + col] = o2t[v] * inv;
    Out[rb + 48 + col] = o3t[v] * inv;
  }
}

// ================================================================ host
extern "C" void kernel_launch(void* const* d_in, const int* in_sizes, int n_in,
                              void* d_out, int out_size, void* d_ws, size_t ws_size,
                              hipStream_t stream) {
  (void)in_sizes; (void)n_in; (void)out_size; (void)ws_size;
  const float* hidden = (const float*)d_in[0];
  const float* past   = (const float*)d_in[1];
  // d_in[2] causal_mask: causality generated in-kernel
  const float* Wq  = (const float*)d_in[3];  const float* bq  = (const float*)d_in[4];
  const float* Wk0 = (const float*)d_in[5];  const float* bk0 = (const float*)d_in[6];
  const float* Wv0 = (const float*)d_in[7];  const float* bv0 = (const float*)d_in[8];
  const float* Wk1 = (const float*)d_in[9];  const float* bk1 = (const float*)d_in[10];
  const float* Wv1 = (const float*)d_in[11]; const float* bv1 = (const float*)d_in[12];
  const float* Wc  = (const float*)d_in[13]; const float* bc  = (const float*)d_in[14];
  const float* Wo  = (const float*)d_in[15]; const float* bo  = (const float*)d_in[16];
  const float* Wg  = (const float*)d_in[17]; const float* bg  = (const float*)d_in[18];
  const float* ln1g = (const float*)d_in[19]; const float* ln1b = (const float*)d_in[20];
  const float* ln2g = (const float*)d_in[21]; const float* ln2b = (const float*)d_in[22];
  const float* Wf1 = (const float*)d_in[23]; const float* bf1 = (const float*)d_in[24];
  const float* Wf2 = (const float*)d_in[25]; const float* bf2 = (const float*)d_in[26];

  float* out_hidden = (float*)d_out;
  float* out_normed = out_hidden + (size_t)BS * DIM;

  char* base = (char*)d_ws;
  size_t off = 0;
  auto alloc = [&](size_t bytes) -> void* {
    void* p = base + off;
    off = (off + bytes + 255) & ~(size_t)255;
    return p;
  };
  const size_t WSZ = (size_t)DIM * DIM;      // 1M
  const size_t BSD = (size_t)BS * DIM;       // 4M
  unsigned short* wq_b  = (unsigned short*)alloc(WSZ * 2);
  unsigned short* wk0_b = (unsigned short*)alloc(WSZ * 2);
  unsigned short* wv0_b = (unsigned short*)alloc(WSZ * 2);
  unsigned short* wk1_b = (unsigned short*)alloc(WSZ * 2);
  unsigned short* wv1_b = (unsigned short*)alloc(WSZ * 2);
  unsigned short* wc_b  = (unsigned short*)alloc(WSZ * 2);
  unsigned short* wo_b  = (unsigned short*)alloc(WSZ * 2);
  unsigned short* wf1_b = (unsigned short*)alloc(WSZ * 4 * 2);
  unsigned short* wf2_b = (unsigned short*)alloc(WSZ * 4 * 2);
  unsigned short* normed_b = (unsigned short*)alloc(BSD * 2);
  unsigned short* qb  = (unsigned short*)alloc(BSD * 2);
  unsigned short* k0b = (unsigned short*)alloc(BSD * 2);
  unsigned short* v0b = (unsigned short*)alloc(BSD * 2);
  unsigned short* pooled_b = (unsigned short*)alloc(BSD * 2); // B*CLEN*DIM == BSD
  unsigned short* cprojb   = (unsigned short*)alloc(BSD * 2);
  unsigned short* k1b = (unsigned short*)alloc(BSD * 2);
  unsigned short* v1b = (unsigned short*)alloc(BSD * 2);
  float* o0f = (float*)alloc(BSD * 4);
  float* o1f = (float*)alloc(BSD * 4);   // contiguous after o0f
  float* cmean = (float*)alloc((size_t)BATCH * DIM * 4);
  float* gatep = (float*)alloc((size_t)BATCH * 2 * 4);
  // aliases (lifetimes disjoint):
  unsigned short* attnb = cprojb;                 // after k1/v1 gemms
  unsigned short* h2b   = pooled_b;               // after cproj gemm
  unsigned short* ffn1b = (unsigned short*)o0f;   // spans o0f+o1f = BS*4D bf16

  const int T = 256;
  auto cv = [&](const float* src, unsigned short* dst, size_t n) {
    cvt_kernel<<<(int)((n + T - 1) / T), T, 0, stream>>>(src, dst, (int)n);
  };
  const dim3 g1024(1024 / 64, BS / 128);     // N=1024 GEMM grid
  const dim3 g4096(4096 / 64, BS / 128);     // N=4096 GEMM grid

  // weights -> bf16
  cv(Wq, wq_b, WSZ);  cv(Wk0, wk0_b, WSZ); cv(Wv0, wv0_b, WSZ);
  cv(Wk1, wk1_b, WSZ); cv(Wv1, wv1_b, WSZ); cv(Wc, wc_b, WSZ);
  cv(Wo, wo_b, WSZ);  cv(Wf1, wf1_b, WSZ * 4); cv(Wf2, wf2_b, WSZ * 4);

  // LN1 (normed is also an output) + gate
  ln_kernel<<<BS, 256, 0, stream>>>(hidden, ln1g, ln1b, out_normed, normed_b);
  colmean_kernel<<<dim3(DIM / 256, BATCH), 256, 0, stream>>>(out_normed, cmean);
  gate_kernel<<<BATCH, 256, 0, stream>>>(cmean, Wg, bg, gatep);

  // projections (N=1024, K=1024, no act, no resid, bf16 out)
  gemm_bf16_kernel<1024, 1024, 0, false, false><<<g1024, 256, 0, stream>>>(
      normed_b, wq_b, bq, qb, nullptr, BS);
  gemm_bf16_kernel<1024, 1024, 0, false, false><<<g1024, 256, 0, stream>>>(
      normed_b, wk0_b, bk0, k0b, nullptr, BS);
  gemm_bf16_kernel<1024, 1024, 0, false, false><<<g1024, 256, 0, stream>>>(
      normed_b, wv0_b, bv0, v0b, nullptr, BS);

  // compressed branch
  pool_kernel<<<(int)((BSD + T - 1) / T), T, 0, stream>>>(past, pooled_b, (int)BSD);
  gemm_bf16_kernel<1024, 1024, 0, false, false><<<g1024, 256, 0, stream>>>(
      pooled_b, wc_b, bc, cprojb, nullptr, BATCH * CLEN);
  gemm_bf16_kernel<1024, 1024, 0, false, false><<<g1024, 256, 0, stream>>>(
      cprojb, wk1_b, bk1, k1b, nullptr, BATCH * CLEN);
  gemm_bf16_kernel<1024, 1024, 0, false, false><<<g1024, 256, 0, stream>>>(
      cprojb, wv1_b, bv1, v1b, nullptr, BATCH * CLEN);

  // attentions (Skv = 1024 both)
  flash_attn_kernel<1><<<dim3(SEQ / 64, BATCH * NH), 128, 0, stream>>>(
      qb, k0b, v0b, o0f);
  flash_attn_kernel<0><<<dim3(SEQ / 64, BATCH * NH), 128, 0, stream>>>(
      qb, k1b, v1b, o1f);

  // gate-combine, output projection (+residual), LN2, FFN (+residual)
  combine_kernel<<<(int)((BSD + T - 1) / T), T, 0, stream>>>(o0f, o1f, gatep, attnb,
                                                             (int)BSD);
  gemm_bf16_kernel<1024, 1024, 0, true, true><<<g1024, 256, 0, stream>>>(
      attnb, wo_b, bo, out_hidden, hidden, BS);
  ln_kernel<<<BS, 256, 0, stream>>>(out_hidden, ln2g, ln2b, nullptr, h2b);
  gemm_bf16_kernel<4096, 1024, 1, false, false><<<g4096, 256, 0, stream>>>(
      h2b, wf1_b, bf1, ffn1b, nullptr, BS);
  gemm_bf16_kernel<1024, 4096, 0, true, true><<<g1024, 256, 0, stream>>>(
      ffn1b, wf2_b, bf2, out_hidden, out_hidden, BS);
}
